// CausalSelfAttention_74809740362128
// MI455X (gfx1250) — matrix operand order repaired and checked by reading, not compiled, not run on hardware
//
#include <hip/hip_runtime.h>

#define B_  4
#define S_  2048
#define DM_ 1024
#define H_  16
#define DK_ 64

#ifndef USE_ASYNC_LDS
#define USE_ASYNC_LDS 1
#endif

typedef __attribute__((ext_vector_type(16))) _Float16 v16h;
typedef __attribute__((ext_vector_type(8)))  _Float16 h8;
typedef __attribute__((ext_vector_type(4)))  _Float16 h4;
typedef __attribute__((ext_vector_type(8)))  float    v8f;

union F16Frag { v16h v; h8 h[2]; };

__device__ __forceinline__ v16h ld_frag(const _Float16* p0, const _Float16* p1) {
    F16Frag u;
    u.h[0] = *(const h8*)p0;
    u.h[1] = *(const h8*)p1;
    return u.v;
}

__device__ __forceinline__ v8f wmma_f16(v16h a, v16h b, v8f c) {
    // D = A(16x32 f16) * B(32x16 f16) + C(16x16 f32)
    return __builtin_amdgcn_wmma_f32_16x16x32_f16(
        /*neg_a=*/false, a, /*neg_b=*/false, b,
        /*c_mod=*/(short)0, c, /*reuse_a=*/false, /*reuse_b=*/false);
}

__device__ __forceinline__ v8f vzero8() {
    v8f z = {0.f,0.f,0.f,0.f,0.f,0.f,0.f,0.f};
    return z;
}

__device__ __forceinline__ void wait_asynccnt0() {
#if USE_ASYNC_LDS
#if __has_builtin(__builtin_amdgcn_s_wait_asynccnt)
    __builtin_amdgcn_s_wait_asynccnt(0);
#else
    asm volatile("s_wait_asynccnt 0x0" ::: "memory");
#endif
#endif
}

// ---------------------------------------------------------------------------
// Projection GEMM: Y = X @ W + b. Output f16 scattered to
//   VTRANS=false: [B, H, S, DK]   (Q, K)
//   VTRANS=true : [B, H, DK, S]   (V, pre-transposed so attention's P@V
//                                  B-fragments are contiguous in global)
// M=8192, N=1024, K=1024. Block tile 128x128, 8 waves, wave tile 64x32.
// ---------------------------------------------------------------------------
template <bool VTRANS>
__global__ void __launch_bounds__(256)
proj_gemm_kernel(const float* __restrict__ X, const float* __restrict__ W,
                 const float* __restrict__ bias, _Float16* __restrict__ dst) {
    __shared__ _Float16 As[128][40];   // [m][k], 40-half stride => 16B-aligned rows
    __shared__ _Float16 Bs[128][40];   // [n][k] (transposed W tile)

    const int t    = threadIdx.x;
    const int bm   = blockIdx.x;       // 64 tiles of 128 rows
    const int bn   = blockIdx.y;       // 8 tiles of 128 cols
    const int wave = t >> 5, lane = t & 31;
    const int lm   = lane & 15, lh = lane >> 4;
    const int wm   = wave >> 2, wn = wave & 3;
    const int rm   = wm * 64, cn = wn * 32;

    v8f acc[4][2];
    for (int i = 0; i < 4; ++i)
        for (int j = 0; j < 2; ++j) acc[i][j] = vzero8();

    float bval[2];
    for (int j = 0; j < 2; ++j) bval[j] = bias[bn * 128 + cn + j * 16 + lm];

    const int tr = t >> 3, tc = (t & 7) * 4;       // A-tile loader coords
    const int wk = t >> 5, wnc = (t & 31) * 4;     // B-tile loader coords

    for (int k0 = 0; k0 < DM_; k0 += 32) {
        // A tile: 128x32 f32 -> f16 LDS
        for (int pp = 0; pp < 4; ++pp) {
            int row = pp * 32 + tr;
            float4 f = *(const float4*)(X + (size_t)(bm * 128 + row) * DM_ + k0 + tc);
            h4 hv = {(_Float16)f.x, (_Float16)f.y, (_Float16)f.z, (_Float16)f.w};
            *(h4*)&As[row][tc] = hv;
        }
        // B tile: 32x128 f32 -> transposed f16 LDS Bs[n][k]
        for (int pp = 0; pp < 4; ++pp) {
            int kk = pp * 8 + wk;
            float4 f = *(const float4*)(W + (size_t)(k0 + kk) * DM_ + bn * 128 + wnc);
            Bs[wnc + 0][kk] = (_Float16)f.x;
            Bs[wnc + 1][kk] = (_Float16)f.y;
            Bs[wnc + 2][kk] = (_Float16)f.z;
            Bs[wnc + 3][kk] = (_Float16)f.w;
        }
        __syncthreads();

        v16h bfr[2];
        for (int j = 0; j < 2; ++j) {
            const _Float16* bp = &Bs[cn + j * 16 + lm][16 * lh];
            bfr[j] = ld_frag(bp, bp + 8);
        }
        for (int i = 0; i < 4; ++i) {
            const _Float16* ap = &As[rm + i * 16 + lm][0];
            v16h af = ld_frag(ap + 16 * lh, ap + 16 * lh + 8);
            for (int j = 0; j < 2; ++j)
                acc[i][j] = wmma_f16(af, bfr[j], acc[i][j]);
        }
        __syncthreads();
    }

    // Epilogue: add bias, scatter f16
    for (int i = 0; i < 4; ++i)
        for (int j = 0; j < 2; ++j) {
            int col = bn * 128 + cn + j * 16 + lm;
            int hh = col >> 6, dd = col & 63;
            for (int r = 0; r < 8; ++r) {
                int row = bm * 128 + rm + i * 16 + r + 8 * lh;
                int bb = row >> 11, tq = row & (S_ - 1);
                float vv = acc[i][j][r] + bval[j];
                size_t bh = (size_t)(bb * H_ + hh);
                if (VTRANS)
                    dst[(bh * DK_ + dd) * S_ + tq] = (_Float16)vv;
                else
                    dst[(bh * S_ + tq) * DK_ + dd] = (_Float16)vv;
            }
        }
}

// ---------------------------------------------------------------------------
// Flash-style causal attention. Grid: (S/128, B*H). 8 waves; each wave owns a
// 16-row query stripe. Key blocks of 64 staged into double-buffered LDS by the
// async-copy engine (one block ahead of compute). V fragments come straight
// from the pre-transposed global Vt. Output -> A2 [B, S, H*DK] f16.
// ---------------------------------------------------------------------------
__global__ void __launch_bounds__(256)
attn_kernel(const _Float16* __restrict__ Qh, const _Float16* __restrict__ Kh,
            const _Float16* __restrict__ Vt, _Float16* __restrict__ A2) {
#if USE_ASYNC_LDS
    __shared__ _Float16 Ksh[2][64][64];    // double-buffered K block [key][d]
#endif
    __shared__ _Float16 plds[8][16][64];   // per-wave P scratch

    const int t    = threadIdx.x;
    const int qblk = blockIdx.x;
    const int bh   = blockIdx.y;
    const int b    = bh >> 4, h = bh & 15;
    const int wave = t >> 5, lane = t & 31;
    const int lm   = lane & 15, lh = lane >> 4;
    const int q0   = qblk * 128 + wave * 16;

    const _Float16* Qb  = Qh + (size_t)bh * S_ * DK_;
    const _Float16* Kb  = Kh + (size_t)bh * S_ * DK_;
    const _Float16* Vtb = Vt + (size_t)bh * DK_ * S_;

    // Q fragments (A-layout), loaded once
    v16h qf[2];
    {
        const _Float16* qp = Qb + (size_t)(q0 + lm) * DK_;
        for (int ks = 0; ks < 2; ++ks)
            qf[ks] = ld_frag(qp + ks * 32 + 16 * lh, qp + ks * 32 + 16 * lh + 8);
    }

    v8f O[4];
    for (int nt = 0; nt < 4; ++nt) O[nt] = vzero8();
    float mrun[8], lrun[8];
    for (int r = 0; r < 8; ++r) { mrun[r] = -3.0e38f; lrun[r] = 0.f; }

    const int nkb = 2 * (qblk + 1);

#if USE_ASYNC_LDS
    const uint32_t ksh_off[2] = {
        (uint32_t)(uintptr_t)&Ksh[0][0][0],   // low 32 bits == LDS byte offset
        (uint32_t)(uintptr_t)&Ksh[1][0][0]
    };
    // Stage key block 0 into buffer 0 (wave 0 drives the async engine).
    if (wave == 0) {
        const _Float16* g0 = Kb;   // block 0 is 8KB contiguous [64][64] f16
        for (int i = 0; i < 16; ++i) {
            uint32_t loff = ksh_off[0] + (uint32_t)(i * 512 + lane * 16);
            const _Float16* g = g0 + i * 256 + lane * 8;
            asm volatile("global_load_async_to_lds_b128 %0, %1, off"
                         :: "v"(loff), "v"(g) : "memory");
        }
    }
#endif

    for (int kb = 0; kb < nkb; ++kb) {
        const int k0  = kb * 64;
#if USE_ASYNC_LDS
        const int buf = kb & 1;
        if (wave == 0) wait_asynccnt0();   // buffer `buf` landed in LDS
        __syncthreads();
        // Kick off staging of the NEXT block into !buf; overlaps compute below.
        if (wave == 0 && (kb + 1) < nkb) {
            const _Float16* g0 = Kb + (size_t)(k0 + 64) * DK_;
            for (int i = 0; i < 16; ++i) {
                uint32_t loff = ksh_off[buf ^ 1] + (uint32_t)(i * 512 + lane * 16);
                const _Float16* g = g0 + i * 256 + lane * 8;
                asm volatile("global_load_async_to_lds_b128 %0, %1, off"
                             :: "v"(loff), "v"(g) : "memory");
            }
        }
#endif
        if (kb + 1 < nkb)   // hint next V block toward the caches
            __builtin_prefetch(Vtb + (size_t)(lane * 2) * S_ + k0 + 64, 0, 1);

        // Scores S = Q K^T
        v8f Sc[4];
        for (int nt = 0; nt < 4; ++nt) Sc[nt] = vzero8();
        for (int ks = 0; ks < 2; ++ks)
            for (int nt = 0; nt < 4; ++nt) {
#if USE_ASYNC_LDS
                const _Float16* kp = &Ksh[buf][nt * 16 + lm][ks * 32 + 16 * lh];
#else
                const _Float16* kp =
                    Kb + (size_t)(k0 + nt * 16 + lm) * DK_ + ks * 32 + 16 * lh;
#endif
                v16h kf = ld_frag(kp, kp + 8);
                Sc[nt] = wmma_f16(qf[ks], kf, Sc[nt]);
            }

        // Causal mask + 1/sqrt(Dk) scale
        for (int nt = 0; nt < 4; ++nt) {
            int kj = k0 + nt * 16 + lm;
            for (int r = 0; r < 8; ++r) {
                int qi = q0 + r + 8 * lh;
                float s = Sc[nt][r] * 0.125f;
                Sc[nt][r] = (kj <= qi) ? s : -1.0e30f;
            }
        }

        // Online softmax (row r lives in VGPR r, cols across 16-lane halves)
        float alpha[8];
        for (int r = 0; r < 8; ++r) {
            float mx = fmaxf(fmaxf(Sc[0][r], Sc[1][r]), fmaxf(Sc[2][r], Sc[3][r]));
            for (int off = 1; off < 16; off <<= 1)
                mx = fmaxf(mx, __shfl_xor(mx, off, 32));
            float mn = fmaxf(mrun[r], mx);
            alpha[r] = __expf(mrun[r] - mn);
            mrun[r] = mn;
            float rs = 0.f;
            for (int nt = 0; nt < 4; ++nt) {
                float e = __expf(Sc[nt][r] - mn);
                Sc[nt][r] = e;
                rs += e;
            }
            for (int off = 1; off < 16; off <<= 1)
                rs += __shfl_xor(rs, off, 32);
            lrun[r] = lrun[r] * alpha[r] + rs;
        }

        // Rescale O; stage P through per-wave LDS for C-layout -> A-layout
        for (int nt = 0; nt < 4; ++nt)
            for (int r = 0; r < 8; ++r) {
                O[nt][r] *= alpha[r];
                plds[wave][r + 8 * lh][nt * 16 + lm] = (_Float16)Sc[nt][r];
            }

        // O += P @ V   (V B-fragments contiguous in pre-transposed global Vt)
        for (int ks = 0; ks < 2; ++ks) {
            const _Float16* pp = &plds[wave][lm][ks * 32];
            v16h pf = ld_frag(pp + 16 * lh, pp + 16 * lh + 8);
            for (int nt = 0; nt < 4; ++nt) {
                const _Float16* vp =
                    Vtb + (size_t)(nt * 16 + lm) * S_ + k0 + ks * 32 + 16 * lh;
                v16h vf = ld_frag(vp, vp + 8);
                O[nt] = wmma_f16(pf, vf, O[nt]);
            }
        }
    }

    // Normalize and store to A2 [B, S, H*DK] f16
    for (int r = 0; r < 8; ++r) {
        float inv = 1.0f / lrun[r];
        int qi = q0 + r + 8 * lh;
        size_t orow = ((size_t)b * S_ + qi) * (H_ * DK_) + h * DK_;
        for (int nt = 0; nt < 4; ++nt)
            A2[orow + nt * 16 + lm] = (_Float16)(O[nt][r] * inv);
    }
}

// ---------------------------------------------------------------------------
// Output GEMM: Y = A2 @ Wo + bo, fp32 out. A fragments direct from global f16.
// ---------------------------------------------------------------------------
__global__ void __launch_bounds__(256)
out_gemm_kernel(const _Float16* __restrict__ A2, const float* __restrict__ Wo,
                const float* __restrict__ bo, float* __restrict__ Y) {
    __shared__ _Float16 Bs[128][40];   // [n][k]

    const int t    = threadIdx.x;
    const int bm   = blockIdx.x;
    const int bn   = blockIdx.y;
    const int wave = t >> 5, lane = t & 31;
    const int lm   = lane & 15, lh = lane >> 4;
    const int wm   = wave >> 2, wn = wave & 3;
    const int rm   = wm * 64, cn = wn * 32;

    v8f acc[4][2];
    for (int i = 0; i < 4; ++i)
        for (int j = 0; j < 2; ++j) acc[i][j] = vzero8();

    float bval[2];
    for (int j = 0; j < 2; ++j) bval[j] = bo[bn * 128 + cn + j * 16 + lm];

    const int wk = t >> 5, wnc = (t & 31) * 4;

    for (int k0 = 0; k0 < DM_; k0 += 32) {
        for (int pp = 0; pp < 4; ++pp) {
            int kk = pp * 8 + wk;
            float4 f = *(const float4*)(Wo + (size_t)(k0 + kk) * DM_ + bn * 128 + wnc);
            Bs[wnc + 0][kk] = (_Float16)f.x;
            Bs[wnc + 1][kk] = (_Float16)f.y;
            Bs[wnc + 2][kk] = (_Float16)f.z;
            Bs[wnc + 3][kk] = (_Float16)f.w;
        }
        __syncthreads();

        v16h bfr[2];
        for (int j = 0; j < 2; ++j) {
            const _Float16* bp = &Bs[cn + j * 16 + lm][16 * lh];
            bfr[j] = ld_frag(bp, bp + 8);
        }
        for (int i = 0; i < 4; ++i) {
            const _Float16* ap =
                A2 + (size_t)(bm * 128 + rm + i * 16 + lm) * DM_ + k0;
            v16h af = ld_frag(ap + 16 * lh, ap + 16 * lh + 8);
            for (int j = 0; j < 2; ++j)
                acc[i][j] = wmma_f16(af, bfr[j], acc[i][j]);
        }
        __syncthreads();
    }

    for (int i = 0; i < 4; ++i)
        for (int j = 0; j < 2; ++j) {
            int col = bn * 128 + cn + j * 16 + lm;
            for (int r = 0; r < 8; ++r) {
                int row = bm * 128 + rm + i * 16 + r + 8 * lh;
                Y[(size_t)row * DM_ + col] = acc[i][j][r] + bval[j];
            }
        }
}

// ---------------------------------------------------------------------------
extern "C" void kernel_launch(void* const* d_in, const int* in_sizes, int n_in,
                              void* d_out, int out_size, void* d_ws, size_t ws_size,
                              hipStream_t stream) {
    (void)in_sizes; (void)n_in; (void)out_size; (void)ws_size;
    const float* q  = (const float*)d_in[0];
    const float* k  = (const float*)d_in[1];
    const float* v  = (const float*)d_in[2];
    const float* Wq = (const float*)d_in[3];
    const float* bq = (const float*)d_in[4];
    const float* Wk = (const float*)d_in[5];
    const float* bk = (const float*)d_in[6];
    const float* Wv = (const float*)d_in[7];
    const float* bv = (const float*)d_in[8];
    const float* Wo = (const float*)d_in[9];
    const float* bo = (const float*)d_in[10];

    _Float16* ws = (_Float16*)d_ws;
    const size_t T = (size_t)B_ * H_ * S_ * DK_;   // 8Mi f16 elements each
    _Float16* Qh = ws;
    _Float16* Kh = ws + T;
    _Float16* Vt = ws + 2 * T;                      // [B, H, DK, S]
    _Float16* A2 = ws + 3 * T;

    dim3 blk(256);
    dim3 gproj(64, 8);
    proj_gemm_kernel<false><<<gproj, blk, 0, stream>>>(q, Wq, bq, Qh);
    proj_gemm_kernel<false><<<gproj, blk, 0, stream>>>(k, Wk, bk, Kh);
    proj_gemm_kernel<true ><<<gproj, blk, 0, stream>>>(v, Wv, bv, Vt);
    attn_kernel<<<dim3(16, 64), blk, 0, stream>>>(Qh, Kh, Vt, A2);
    out_gemm_kernel<<<gproj, blk, 0, stream>>>(A2, Wo, bo, (float*)d_out);
}
